// TemporalAttention_70239895159320
// MI455X (gfx1250) — compile-verified, matrix operand order checked by reading, not run on hardware
//
#include <hip/hip_runtime.h>
#include <math.h>

#define B_  32
#define T_  24
#define N_  325
#define D_  64
#define H_  8
#define DH_ 8
#define MASK_VAL (-32767.0f)
#define INV_SQRT_DH 0.3535533905932738f

// packed-weight fragment bases (one fragment = 32 lanes * 16 f16 = 1 KB)
#define FQ_   0
#define FK_   16
#define FV_   32
#define FO1_  48
#define FO2_  56
#define NFRAG_ 64
#define PW_BYTES ((size_t)NFRAG_ * 512 * 2)   // 64 KB, L2-resident

typedef __attribute__((ext_vector_type(16))) _Float16 v16h;
typedef __attribute__((ext_vector_type(8)))  _Float16 v8h;
typedef __attribute__((ext_vector_type(8)))  float    v8f;

#define HPAD 136   // f16 h-slab row stride (272B rows, 16B aligned)
#define QPAD 68    // f32 slab row stride   (272B rows, 16B aligned)
#define APAD 72    // f16 A-staging row stride (144B rows, 16B aligned)

// ---------------------------------------------------------------------------
// WMMA fragment layouts per CDNA5 ISA §7.12.2 (wave32):
// A/B 16x32 f16: lane l (row=l&15, half=l>>4):
//   elem i<8  -> K = 8*half + i ;  elem i>=8 -> K = 16 + 8*half + (i-8)
// C/D f32: lane l holds rows m = j + 8*(l>>4) (j=0..7), column n = l&15.
// ---------------------------------------------------------------------------

// One-time weight pre-pack: f32 (K x 64) row-major -> f16 B-fragments.
__global__ void __launch_bounds__(256)
pack_weights_kernel(const float* __restrict__ Wq, const float* __restrict__ Wk,
                    const float* __restrict__ Wv, const float* __restrict__ Wo1,
                    const float* __restrict__ Wo2, _Float16* __restrict__ pw) {
    const int idx = blockIdx.x * blockDim.x + threadIdx.x;   // 0 .. 2047
    if (idx >= NFRAG_ * 32) return;
    const int f    = idx >> 5;
    const int lane = idx & 31;
    const float* W; int fl;
    if      (f < FK_)  { W = Wq;  fl = f - FQ_;  }
    else if (f < FV_)  { W = Wk;  fl = f - FK_;  }
    else if (f < FO1_) { W = Wv;  fl = f - FV_;  }
    else if (f < FO2_) { W = Wo1; fl = f - FO1_; }
    else               { W = Wo2; fl = f - FO2_; }
    const int kt  = fl >> 2, nt = fl & 3;
    const int row = lane & 15, half = lane >> 4;
    const int n   = nt * 16 + row;
    const int ko  = kt * 32;
    _Float16* d = pw + ((size_t)f << 9) + (lane << 4);
#pragma unroll
    for (int i = 0; i < 8; ++i) {
        d[i]     = (_Float16)W[(size_t)(ko + 8 * half + i) * D_ + n];
        d[i + 8] = (_Float16)W[(size_t)(ko + 16 + 8 * half + i) * D_ + n];
    }
}

// A-fragment gathered from an f16 LDS row: two ds_load_b128, no converts.
__device__ inline v16h a_frag_lds(const _Float16* __restrict__ rowbase, int half) {
    const v8h g0 = *(const v8h*)(rowbase + 8 * half);
    const v8h g1 = *(const v8h*)(rowbase + 16 + 8 * half);
    v16h a;
#pragma unroll
    for (int i = 0; i < 8; ++i) { a[i] = g0[i]; a[i + 8] = g1[i]; }
    return a;
}

// B-fragment: one 32B vector load from the packed buffer.
__device__ inline v16h load_b_frag(const _Float16* __restrict__ pw, int f, int lane) {
    return *(const v16h*)(pw + ((size_t)f << 9) + ((size_t)lane << 4));
}

// ---------------------------------------------------------------------------
// Fully fused kernel: one block per (b, n). QKV -> attention -> MLP entirely
// in LDS; HBM traffic = x/ste reads + out writes only (roofline minimum).
// 8 waves: tile job (rt, nt) = (wave>>2, wave&3) covers the 32x64 output.
// All staging buffers are 32 rows so padded rows (24..31, garbage but
// row-confined) can be written unconditionally -> no exec-mask branches.
// ---------------------------------------------------------------------------
__global__ void __launch_bounds__(256)
fused_kernel(const float* __restrict__ x, const float* __restrict__ ste,
             const _Float16* __restrict__ pw,
             const float* __restrict__ bq, const float* __restrict__ bk,
             const float* __restrict__ bv,
             const float* __restrict__ bo1, const float* __restrict__ bo2,
             float* __restrict__ out) {
    __shared__ __align__(16) _Float16 hA[32][HPAD];   // concat(x,ste) rows, f16
    __shared__ __align__(16) float    sQ[32][QPAD];
    __shared__ __align__(16) float    sK[32][QPAD];
    __shared__ __align__(16) float    sV[32][QPAD];
    __shared__ __align__(16) _Float16 aoh[32][APAD];  // attention out, f16 A rows
    __shared__ __align__(16) _Float16 uh[32][APAD];   // MLP hidden, f16 A rows
    __shared__ __align__(16) float    uo[32][QPAD];   // final f32 tile

    const int tid = threadIdx.x;
    const int b = blockIdx.x / N_;
    const int n = blockIdx.x % N_;
    const size_t rowbase = ((size_t)b * T_ * N_ + n) * D_;   // + t*N_*D_ per row

    // ---- phase 0: h = concat(x, ste) -> f16 LDS (24 rows x 128)
    for (int i = tid; i < T_ * 16; i += 256) {               // 384 8-half chunks
        const int t = i >> 4, c8 = i & 15;
        const float* src = ((c8 < 8) ? x : ste) + rowbase + (size_t)t * N_ * D_
                         + (c8 & 7) * 8;
        const float4 f0 = *(const float4*)src;
        const float4 f1 = *(const float4*)(src + 4);
        v8h hv;
        hv[0] = (_Float16)f0.x; hv[1] = (_Float16)f0.y;
        hv[2] = (_Float16)f0.z; hv[3] = (_Float16)f0.w;
        hv[4] = (_Float16)f1.x; hv[5] = (_Float16)f1.y;
        hv[6] = (_Float16)f1.z; hv[7] = (_Float16)f1.w;
        *(v8h*)&hA[t][c8 * 8] = hv;
    }
    __syncthreads();

    const int lane = tid & 31, wave = tid >> 5;
    const int row  = lane & 15, half = lane >> 4;
    const int rt   = wave >> 2, nt = wave & 3;               // tile job
    const int ncol = nt * 16 + row;
    const int m0   = rt * 16 + 8 * half;                     // first C/D row

    // ---- phase 1: QKV projection (M=32 padded; rows >=24 garbage, unmasked)
    {
        v16h a[4];
#pragma unroll
        for (int kk = 0; kk < 4; ++kk)
            a[kk] = a_frag_lds(&hA[rt * 16 + row][kk * 32], half);

        const int    fb[3]   = {FQ_, FK_, FV_};
        const float* bias[3] = {bq, bk, bv};
        float (*dstS[3])[QPAD] = {sQ, sK, sV};
#pragma unroll
        for (int o = 0; o < 3; ++o) {
            v8f c = {};
#pragma unroll
            for (int kk = 0; kk < 4; ++kk) {
                v16h bf = load_b_frag(pw, fb[o] + kk * 4 + nt, lane);
                c = __builtin_amdgcn_wmma_f32_16x16x32_f16(
                        false, a[kk], false, bf, (short)0, c, false, false);
            }
            const float bb = bias[o][ncol];
#pragma unroll
            for (int j = 0; j < 8; ++j)
                dstS[o][m0 + j][ncol] = fmaxf(c[j] + bb, 0.0f);
        }
    }
    __syncthreads();

    // ---- phase 2: causal attention; threads 0..191 = (head, t), full lanes
    if (tid < H_ * T_) {
        const int head = tid / T_;
        const int t    = tid % T_;
        const int hd   = head * DH_;
        float qv[DH_];
#pragma unroll
        for (int d = 0; d < DH_; ++d) qv[d] = sQ[t][hd + d];

        float sc[T_];
        float mx = MASK_VAL;
#pragma unroll
        for (int s = 0; s < T_; ++s) {
            float acc = 0.0f;
#pragma unroll
            for (int d = 0; d < DH_; ++d) acc += qv[d] * sK[s][hd + d];
            acc *= INV_SQRT_DH;
            sc[s] = (s <= t) ? acc : MASK_VAL;
            mx = fmaxf(mx, sc[s]);
        }
        float sum = 0.0f;
#pragma unroll
        for (int s = 0; s < T_; ++s) {
            sc[s] = __expf(sc[s] - mx);
            sum += sc[s];
        }
        const float inv = 1.0f / sum;
        float acc[DH_] = {};
#pragma unroll
        for (int s = 0; s < T_; ++s) {
            const float pp = sc[s] * inv;
#pragma unroll
            for (int d = 0; d < DH_; ++d) acc[d] += pp * sV[s][hd + d];
        }
        v8h ov;
#pragma unroll
        for (int d = 0; d < DH_; ++d) ov[d] = (_Float16)acc[d];
        *(v8h*)&aoh[t][hd] = ov;              // f16 A-staging row
    }
    __syncthreads();

    // ---- phase 3: MLP GEMM1: u = relu(aoh @ Wo1 + bo1)
    {
        v16h a[2];
#pragma unroll
        for (int kk = 0; kk < 2; ++kk)
            a[kk] = a_frag_lds(&aoh[rt * 16 + row][kk * 32], half);
        v8f c = {};
#pragma unroll
        for (int kk = 0; kk < 2; ++kk) {
            v16h bf = load_b_frag(pw, FO1_ + kk * 4 + nt, lane);
            c = __builtin_amdgcn_wmma_f32_16x16x32_f16(
                    false, a[kk], false, bf, (short)0, c, false, false);
        }
        const float bb = bo1[ncol];
#pragma unroll
        for (int j = 0; j < 8; ++j)
            uh[m0 + j][ncol] = (_Float16)fmaxf(c[j] + bb, 0.0f);
    }
    __syncthreads();

    // ---- phase 4: MLP GEMM2: out = u @ Wo2 + bo2
    {
        v16h a[2];
#pragma unroll
        for (int kk = 0; kk < 2; ++kk)
            a[kk] = a_frag_lds(&uh[rt * 16 + row][kk * 32], half);
        v8f c = {};
#pragma unroll
        for (int kk = 0; kk < 2; ++kk) {
            v16h bf = load_b_frag(pw, FO2_ + kk * 4 + nt, lane);
            c = __builtin_amdgcn_wmma_f32_16x16x32_f16(
                    false, a[kk], false, bf, (short)0, c, false, false);
        }
        const float bb = bo2[ncol];
#pragma unroll
        for (int j = 0; j < 8; ++j)
            uo[m0 + j][ncol] = c[j] + bb;
    }
    __syncthreads();

    // ---- phase 5: store the 24 valid rows, coalesced float4
    for (int i = tid; i < T_ * 16; i += 256) {               // 384 float4 chunks
        const int t = i >> 4, c4 = i & 15;
        *(float4*)(out + rowbase + (size_t)t * N_ * D_ + c4 * 4) =
            *(const float4*)&uo[t][c4 * 4];
    }
}

// ---------------------------------------------------------------------------
extern "C" void kernel_launch(void* const* d_in, const int* in_sizes, int n_in,
                              void* d_out, int out_size, void* d_ws, size_t ws_size,
                              hipStream_t stream) {
    const float* x   = (const float*)d_in[0];
    const float* ste = (const float*)d_in[1];
    const float* Wq  = (const float*)d_in[2];
    const float* bq  = (const float*)d_in[3];
    const float* Wk  = (const float*)d_in[4];
    const float* bk  = (const float*)d_in[5];
    const float* Wv  = (const float*)d_in[6];
    const float* bv  = (const float*)d_in[7];
    const float* Wo1 = (const float*)d_in[8];
    const float* bo1 = (const float*)d_in[9];
    const float* Wo2 = (const float*)d_in[10];
    const float* bo2 = (const float*)d_in[11];

    _Float16* pw = (_Float16*)d_ws;          // only 64 KB of workspace needed

    pack_weights_kernel<<<(NFRAG_ * 32 + 255) / 256, 256, 0, stream>>>(
        Wq, Wk, Wv, Wo1, Wo2, pw);
    fused_kernel<<<B_ * N_, 256, 0, stream>>>(
        x, ste, pw, bq, bk, bv, bo1, bo2, (float*)d_out);
}
